// MapAttentionEncoder_21646635172633
// MI455X (gfx1250) — compile-verified, matrix operand order checked by reading, not run on hardware
//
#include <hip/hip_runtime.h>
#include <hip/hip_bf16.h>
#include <math.h>

typedef _Float16 h16;
typedef __attribute__((ext_vector_type(16))) _Float16 v16h;
typedef __attribute__((ext_vector_type(8)))  float    v8f;

// ---------------------------------------------------------------------------
// WMMA 16x16x32 f16 (CDNA5 V_WMMA_F32_16X16X32_F16, wave32).
// Operand layouts per CDNA5 ISA 7.12.2:
//  A (16x32 f16): lane = m + 16*kg, half h -> k = (h&7) + 8*kg + 16*(h>>3)
//  B (32x16 f16): same with N per lane:  lane = n + 16*kg, same k pattern
//  C/D (16x16 f32): vgpr r -> M = r + 8*(lane>>4), N = lane&15
// Inverse maps (used to produce packed operands): kg=(k>>3)&1,
//  h = (k&7) | (((k>>4)&1)<<3).
// Packed operand storage: tile = 32 lanes x 16 halfs, lane-major (32B/lane),
// so a B/A fetch is one aligned v16h load (2 x ds_load_b128 / global_load_b128).
// ---------------------------------------------------------------------------
__device__ __forceinline__ v8f wmma_f16(v16h a, v16h b, v8f c) {
  return __builtin_amdgcn_wmma_f32_16x16x32_f16(false, a, false, b, (short)0, c, false, false);
}

__device__ __forceinline__ v16h load_A16x32(const h16* __restrict__ src, int ld,
                                            int r0, int c0, int rMax, int cMax) {
  const int lane = threadIdx.x & 31;
  const int m  = lane & 15;
  const int kg = lane >> 4;
  v16h a;
#pragma unroll
  for (int h = 0; h < 16; ++h) {
    int k = (h & 7) + 8 * kg + ((h >> 3) << 4);
    int r = r0 + m, c = c0 + k;
    a[h] = (r < rMax && c < cMax) ? src[(size_t)r * ld + c] : (h16)0.f;
  }
  return a;
}

__device__ __forceinline__ v16h load_packed(const h16* __restrict__ base, int tile) {
  const int lane = threadIdx.x & 31;
  return *(const v16h*)(base + ((size_t)tile << 9) + (lane << 4));
}

__device__ __forceinline__ void store_C_f32(float* __restrict__ dst, int ld,
                                            int r0, int c0, v8f c, int rMax, float scale) {
  const int lane = threadIdx.x & 31;
  const int n  = lane & 15;
  const int mb = (lane >> 4) * 8;
#pragma unroll
  for (int r = 0; r < 8; ++r) {
    int m = r0 + mb + r;
    if (m < rMax) dst[(size_t)m * ld + c0 + n] = c[r] * scale;
  }
}

// Pack a 64x64 f32 weight into 8 B-operand tiles (tile = nt*2+kt) of f16.
__device__ __forceinline__ void pack_W64(const float* __restrict__ W, h16* __restrict__ sWp) {
  const int lane = threadIdx.x & 31;
  for (int i = lane; i < 4096; i += 32) {
    int tile = i >> 9;
    int nt = tile >> 1, kt = tile & 1;
    int l = (i >> 4) & 31, h = i & 15;
    int k = kt * 32 + (h & 7) + 8 * (l >> 4) + ((h >> 3) << 4);
    int n = nt * 16 + (l & 15);
    sWp[i] = (h16)W[k * 64 + n];
  }
}

// (16 x 64) @ (64 x 64, packed) + bias -> f16 dst (row-major, ld 64)
__device__ __forceinline__ void gemm_16x64x64_h16p(const h16* __restrict__ sIn,
                                                   const h16* __restrict__ sWp,
                                                   const float* __restrict__ bias,
                                                   h16* __restrict__ dst) {
  const int lane = threadIdx.x & 31;
#pragma unroll
  for (int nt = 0; nt < 4; ++nt) {
    v8f c = {};
#pragma unroll
    for (int kt = 0; kt < 2; ++kt) {
      v16h a  = load_A16x32(sIn, 64, 0, kt * 32, 16, 64);
      v16h bb = load_packed(sWp, (nt << 1) | kt);
      c = wmma_f16(a, bb, c);
    }
    const int n  = (lane & 15) + nt * 16;
    const int mb = (lane >> 4) * 8;
#pragma unroll
    for (int r = 0; r < 8; ++r)
      dst[(size_t)(mb + r) * 64 + n] = (h16)(c[r] + bias[n]);
  }
}

// ---------------------------------------------------------------------------
// Kernel 1/2: local graph encoder (MLP->LN->ReLU x2, column max, duplicated).
// One wave per node matrix. Rows padded to 32, input K padded to 32 with zeros.
// ---------------------------------------------------------------------------
__global__ __launch_bounds__(32) void k_local_graph(
    const float* __restrict__ xs, int blockStride, int inCols, int nRows,
    const float* __restrict__ W1, const float* __restrict__ b1,
    const float* __restrict__ g1, const float* __restrict__ be1,
    const float* __restrict__ W2, const float* __restrict__ b2,
    const float* __restrict__ g2, const float* __restrict__ be2,
    float* __restrict__ out /* +blockIdx.x*64 */) {
  const int lane = threadIdx.x;
  __shared__ __align__(32) h16 sX[32 * 32];
  __shared__ __align__(32) h16 sWp[2 * 512];  // 2 packed B tiles (K=32)
  __shared__ float sH[32 * 33];

  const float* x = xs + (size_t)blockIdx.x * blockStride;

  for (int i = lane; i < 32 * 32; i += 32) sX[i] = (h16)0.f;
  __syncthreads();
  for (int i = lane; i < nRows * inCols; i += 32) {
    int r = i / inCols, c = i % inCols;
    sX[r * 32 + c] = (h16)x[i];
  }
  // pack W1 (inCols x 32, zero-padded rows) into B-operand layout
  for (int i = lane; i < 1024; i += 32) {
    int nt = i >> 9, l = (i >> 4) & 31, h = i & 15;
    int k = (h & 7) + 8 * (l >> 4) + ((h >> 3) << 4);
    int n = nt * 16 + (l & 15);
    sWp[i] = (k < inCols) ? (h16)W1[k * 32 + n] : (h16)0.f;
  }
  __syncthreads();

  // H1 = X @ W1
#pragma unroll
  for (int mt = 0; mt < 2; ++mt)
#pragma unroll
    for (int nt = 0; nt < 2; ++nt) {
      v16h a  = load_A16x32(sX, 32, mt * 16, 0, 32, 32);
      v16h bb = load_packed(sWp, nt);
      v8f  c  = {};
      c = wmma_f16(a, bb, c);
      store_C_f32(sH, 33, mt * 16, nt * 16, c, 32, 1.f);
    }
  __syncthreads();

  // LN + ReLU, write f16 back into sX; also stage packed W2
  if (lane < nRows) {
    float hv[32];
    float mu = 0.f;
#pragma unroll
    for (int c = 0; c < 32; ++c) { float v = sH[lane * 33 + c] + b1[c]; hv[c] = v; mu += v; }
    mu *= (1.f / 32.f);
    float m2 = 0.f;
#pragma unroll
    for (int c = 0; c < 32; ++c) { float d = hv[c] - mu; m2 += d * d; }
    float rinv = rsqrtf(m2 * (1.f / 32.f) + 1e-5f);
#pragma unroll
    for (int c = 0; c < 32; ++c) {
      float v = (hv[c] - mu) * rinv * g1[c] + be1[c];
      sX[lane * 32 + c] = (h16)fmaxf(v, 0.f);
    }
  }
  for (int i = lane; i < 1024; i += 32) {
    int nt = i >> 9, l = (i >> 4) & 31, h = i & 15;
    int k = (h & 7) + 8 * (l >> 4) + ((h >> 3) << 4);
    int n = nt * 16 + (l & 15);
    sWp[i] = (h16)W2[k * 32 + n];
  }
  __syncthreads();

  // H2 = H1' @ W2
#pragma unroll
  for (int mt = 0; mt < 2; ++mt)
#pragma unroll
    for (int nt = 0; nt < 2; ++nt) {
      v16h a  = load_A16x32(sX, 32, mt * 16, 0, 32, 32);
      v16h bb = load_packed(sWp, nt);
      v8f  c  = {};
      c = wmma_f16(a, bb, c);
      store_C_f32(sH, 33, mt * 16, nt * 16, c, 32, 1.f);
    }
  __syncthreads();

  if (lane < nRows) {
    float hv[32];
    float mu = 0.f;
#pragma unroll
    for (int c = 0; c < 32; ++c) { float v = sH[lane * 33 + c] + b2[c]; hv[c] = v; mu += v; }
    mu *= (1.f / 32.f);
    float m2 = 0.f;
#pragma unroll
    for (int c = 0; c < 32; ++c) { float d = hv[c] - mu; m2 += d * d; }
    float rinv = rsqrtf(m2 * (1.f / 32.f) + 1e-5f);
#pragma unroll
    for (int c = 0; c < 32; ++c) {
      float v = (hv[c] - mu) * rinv * g2[c] + be2[c];
      sH[lane * 33 + c] = fmaxf(v, 0.f);
    }
  }
  __syncthreads();

  // Column-max over valid rows; nb-half equals this exactly (ReLU>=0, P>=2)
  {
    float mx = 0.f;
    for (int r = 0; r < nRows; ++r) mx = fmaxf(mx, sH[r * 33 + lane]);
    float* o = out + (size_t)blockIdx.x * 64;
    o[lane] = mx;
    o[lane + 32] = mx;
  }
}

// ---------------------------------------------------------------------------
// Kernel 3: Q/K/V projections for attention 0 (kv = [target; mapEmb]).
// grid (B, 9). Results are scattered to *packed WMMA-operand* global layouts:
//   Kp: B^T tiles for scores   (tile = keytile*2 + featgroup,  18 tiles/b)
//   Vp: B   tiles for attn@V   (tile = feattile*5 + keytile32, 20 tiles/b)
//   Qp: A   tiles              (tile = qtile*2  + featgroup,   16 tiles/b)
// ---------------------------------------------------------------------------
__global__ __launch_bounds__(32) void k_proj0(
    const float* __restrict__ mapEmb, const float* __restrict__ target,
    const float* __restrict__ Wq, const float* __restrict__ bq,
    const float* __restrict__ Wk, const float* __restrict__ bk,
    const float* __restrict__ Wv, const float* __restrict__ bv,
    h16* __restrict__ Qp, h16* __restrict__ Kp, h16* __restrict__ Vp) {
  const int b = blockIdx.x, t = blockIdx.y, lane = threadIdx.x;
  __shared__ __align__(32) h16 sIn[16 * 64];
  __shared__ __align__(32) h16 sWp[64 * 64];

  for (int i = lane; i < 16 * 64; i += 32) {
    int r = 16 * t + (i >> 6), c = i & 63;
    float v = 0.f;
    if (r == 0)       v = target[(size_t)b * 64 + c];
    else if (r < 129) v = mapEmb[((size_t)b * 128 + (r - 1)) * 64 + c];
    sIn[i] = (h16)v;
  }
  pack_W64(Wk, sWp);
  __syncthreads();

  // ---- K -> packed B^T tiles (B(k=feat, n=key)) ----
  h16* KpB = Kp + (size_t)b * 18 * 512;
#pragma unroll
  for (int nt = 0; nt < 4; ++nt) {
    v8f c = {};
#pragma unroll
    for (int kt = 0; kt < 2; ++kt) {
      v16h a  = load_A16x32(sIn, 64, 0, kt * 32, 16, 64);
      v16h bb = load_packed(sWp, (nt << 1) | kt);
      c = wmma_f16(a, bb, c);
    }
    const int feat = nt * 16 + (lane & 15);
    const int mb   = (lane >> 4) * 8;
    const int ktF  = feat >> 5, kl = feat & 31;
    const int hh   = (kl & 7) | (((kl >> 4) & 1) << 3);
    const int lbit = 16 * ((kl >> 3) & 1);
#pragma unroll
    for (int r = 0; r < 8; ++r) {
      int keyl = mb + r;  // key = 16*t + keyl, keytile = t
      float val = c[r] + bk[feat];
      KpB[(((size_t)t * 2 + ktF) << 9) + (((keyl & 15) + lbit) << 4) + hh] = (h16)val;
    }
  }
  __syncthreads();

  // ---- V -> packed B tiles (B(k=key, n=feat)) ----
  pack_W64(Wv, sWp);
  __syncthreads();
  h16* VpB = Vp + (size_t)b * 20 * 512;
#pragma unroll
  for (int nt = 0; nt < 4; ++nt) {
    v8f c = {};
#pragma unroll
    for (int kt = 0; kt < 2; ++kt) {
      v16h a  = load_A16x32(sIn, 64, 0, kt * 32, 16, 64);
      v16h bb = load_packed(sWp, (nt << 1) | kt);
      c = wmma_f16(a, bb, c);
    }
    const int feat = nt * 16 + (lane & 15);
    const int mb   = (lane >> 4) * 8;
#pragma unroll
    for (int r = 0; r < 8; ++r) {
      int key = 16 * t + mb + r;
      float val = c[r] + bv[feat];
      int ktK = key >> 5, kl = key & 31;
      int hh  = (kl & 7) | (((kl >> 4) & 1) << 3);
      int ll  = (feat & 15) + 16 * ((kl >> 3) & 1);
      VpB[(((size_t)nt * 5 + ktK) << 9) + (ll << 4) + hh] = (h16)val;
    }
  }
  // zero-fill padded keys 144..159 (tile ktK=4) so the K=160 accumulation is exact
  if (t == 8) {
    for (int i = lane; i < 4 * 16 * 16; i += 32) {
      int nt = i >> 8;
      int keyl = 16 + ((i >> 4) & 15);
      int f = i & 15;
      int hh = (keyl & 7) | (((keyl >> 4) & 1) << 3);
      int ll = f + 16 * ((keyl >> 3) & 1);
      VpB[(((size_t)nt * 5 + 4) << 9) + (ll << 4) + hh] = (h16)0.f;
    }
  }
  __syncthreads();

  // ---- Q -> packed A tiles ----
  if (t < 8) {
    for (int i = lane; i < 16 * 64; i += 32) {
      int r = 16 * t + (i >> 6), c = i & 63;
      sIn[i] = (h16)mapEmb[((size_t)b * 128 + r) * 64 + c];
    }
    pack_W64(Wq, sWp);
    __syncthreads();
    h16* QpB = Qp + ((size_t)b * 8 + t) * 1024;
#pragma unroll
    for (int nt = 0; nt < 4; ++nt) {
      v8f c = {};
#pragma unroll
      for (int kt = 0; kt < 2; ++kt) {
        v16h a  = load_A16x32(sIn, 64, 0, kt * 32, 16, 64);
        v16h bb = load_packed(sWp, (nt << 1) | kt);
        c = wmma_f16(a, bb, c);
      }
      const int feat = nt * 16 + (lane & 15);
      const int mb   = (lane >> 4) * 8;
      const int ktF  = feat >> 5, kl = feat & 31;
      const int hh   = (kl & 7) | (((kl >> 4) & 1) << 3);
      const int lbit = 16 * ((kl >> 3) & 1);
#pragma unroll
      for (int r = 0; r < 8; ++r) {
        float val = c[r] + bq[feat];
        QpB[((size_t)ktF << 9) + (((mb + r) + lbit) << 4) + hh] = (h16)val;
      }
    }
  }
}

// ---------------------------------------------------------------------------
// Kernel 4: attention 0 core, one 16-row Q tile per wave. grid (B, 8).
// All WMMA B operands are pre-packed -> pure ds_load_b128 / global_load_b128.
// ---------------------------------------------------------------------------
__global__ __launch_bounds__(32) void k_attn0(
    const h16* __restrict__ Qp, const h16* __restrict__ Kp, const h16* __restrict__ Vp,
    const float* __restrict__ Wo, const float* __restrict__ bo,
    const float* __restrict__ mapEmb, float* __restrict__ mapEmb2,
    h16* __restrict__ m2h, float* __restrict__ outp) {
  const int b = blockIdx.x, qt = blockIdx.y, lane = threadIdx.x;
  __shared__ __align__(32) h16 sKp[18 * 512];
  __shared__ __align__(32) h16 sVp[20 * 512];
  __shared__ float sSc[16 * 144];
  __shared__ __align__(32) h16 sAt[16 * 160];
  __shared__ __align__(32) h16 sCh[16 * 64];
  __shared__ __align__(32) h16 sWo[64 * 64];

  {
    const uint4* kf = (const uint4*)(Kp + (size_t)b * 18 * 512);
    const uint4* vf = (const uint4*)(Vp + (size_t)b * 20 * 512);
    uint4* ks = (uint4*)sKp;
    uint4* vs = (uint4*)sVp;
    for (int i = lane; i < (18 * 512) / 8; i += 32) ks[i] = kf[i];
    for (int i = lane; i < (20 * 512) / 8; i += 32) vs[i] = vf[i];
  }
  pack_W64(Wo, sWo);
  for (int i = lane; i < 16 * 160; i += 32) sAt[i] = (h16)0.f;
  __syncthreads();

  // Q tile A-operands: direct packed loads from global
  const h16* QpB = Qp + ((size_t)b * 8 + qt) * 1024;
  v16h aQ0 = *(const v16h*)(QpB + (lane << 4));
  v16h aQ1 = *(const v16h*)(QpB + 512 + (lane << 4));

  // scores = Q K^T / 8
#pragma unroll
  for (int nt = 0; nt < 9; ++nt) {
    v8f c = {};
    c = wmma_f16(aQ0, load_packed(sKp, nt * 2 + 0), c);
    c = wmma_f16(aQ1, load_packed(sKp, nt * 2 + 1), c);
    store_C_f32(sSc, 144, 0, nt * 16, c, 16, 0.125f);
  }
  __syncthreads();

  // masked softmax over 129 keys (one row per lane 0..15)
  if (lane < 16) {
    float* row = sSc + lane * 144;
    float mx = -1e30f;
    for (int j = 0; j < 129; ++j) mx = fmaxf(mx, row[j]);
    float den = 0.f;
    for (int j = 0; j < 129; ++j) { float e = __expf(row[j] - mx); row[j] = e; den += e; }
    float rden = 1.f / den;
    for (int j = 0; j < 129; ++j) sAt[lane * 160 + j] = (h16)(row[j] * rden);
  }
  __syncthreads();

  // ctx = attn @ V  (K padded to 160; padded V keys are zero-filled)
#pragma unroll
  for (int nt = 0; nt < 4; ++nt) {
    v8f c = {};
#pragma unroll
    for (int kt = 0; kt < 5; ++kt) {
      v16h a  = load_A16x32(sAt, 160, 0, kt * 32, 16, 160);
      v16h bb = load_packed(sVp, nt * 5 + kt);
      c = wmma_f16(a, bb, c);
    }
    const int n  = (lane & 15) + nt * 16;
    const int mb = (lane >> 4) * 8;
#pragma unroll
    for (int r = 0; r < 8; ++r) sCh[(size_t)(mb + r) * 64 + n] = (h16)c[r];
  }
  __syncthreads();

  // out = ctx @ Wo + bo + residual(mapEmb)
  const size_t rowBase = (size_t)b * 128 + (size_t)qt * 16;
#pragma unroll
  for (int nt = 0; nt < 4; ++nt) {
    v8f c = {};
#pragma unroll
    for (int kt = 0; kt < 2; ++kt) {
      v16h a  = load_A16x32(sCh, 64, 0, kt * 32, 16, 64);
      v16h bb = load_packed(sWo, (nt << 1) | kt);
      c = wmma_f16(a, bb, c);
    }
    const int n  = (lane & 15) + nt * 16;
    const int mb = (lane >> 4) * 8;
#pragma unroll
    for (int r = 0; r < 8; ++r) {
      int m = mb + r;
      size_t g = (rowBase + m) * 64 + n;
      float val = c[r] + bo[n] + mapEmb[g];
      mapEmb2[g] = val;
      m2h[g]     = (h16)val;
      outp[((size_t)b * 129 + 1 + (size_t)qt * 16 + m) * 64 + n] = val;
    }
  }
}

// ---------------------------------------------------------------------------
// Kernel 5: K2/V2 projections of mapEmb2 for t_m2a. grid (B, 8).
// ---------------------------------------------------------------------------
__global__ __launch_bounds__(32) void k_proj2(
    const h16* __restrict__ m2h,
    const float* __restrict__ Wk2, const float* __restrict__ bk2,
    const float* __restrict__ Wv2, const float* __restrict__ bv2,
    h16* __restrict__ K2h, h16* __restrict__ V2h) {
  const int b = blockIdx.x, t = blockIdx.y, lane = threadIdx.x;
  __shared__ __align__(32) h16 sIn[16 * 64];
  __shared__ __align__(32) h16 sWp[64 * 64];

  {
    const uint4* src = (const uint4*)(m2h + ((size_t)b * 128 + 16 * t) * 64);
    uint4* dst = (uint4*)sIn;
    for (int i = lane; i < (16 * 64) / 8; i += 32) dst[i] = src[i];
  }
  pack_W64(Wk2, sWp);
  __syncthreads();
  gemm_16x64x64_h16p(sIn, sWp, bk2, K2h + ((size_t)b * 128 + 16 * t) * 64);
  __syncthreads();
  pack_W64(Wv2, sWp);
  __syncthreads();
  gemm_16x64x64_h16p(sIn, sWp, bv2, V2h + ((size_t)b * 128 + 16 * t) * 64);
}

// ---------------------------------------------------------------------------
// Kernel 6: target row. t_a collapses (1 key -> attn=1); t_m2a is a gemv
// attention over 128 keys. One wave per batch.
// ---------------------------------------------------------------------------
__global__ __launch_bounds__(32) void k_target(
    const float* __restrict__ target,
    const h16* __restrict__ K2h, const h16* __restrict__ V2h,
    const float* __restrict__ Wv1, const float* __restrict__ bv1,
    const float* __restrict__ Wo1, const float* __restrict__ bo1,
    const float* __restrict__ Wq2, const float* __restrict__ bq2,
    const float* __restrict__ Wo2, const float* __restrict__ bo2,
    float* __restrict__ outp) {
  const int b = blockIdx.x, lane = threadIdx.x;
  __shared__ float st[64], tv[64], qv[64], sc[128], ctx[64];

  st[lane]      = target[(size_t)b * 64 + lane];
  st[lane + 32] = target[(size_t)b * 64 + lane + 32];
  __syncthreads();

  for (int cc = lane; cc < 64; cc += 32) {
    float s1 = bv1[cc], s2 = bq2[cc];
    for (int d = 0; d < 64; ++d) {
      s1 += st[d] * Wv1[d * 64 + cc];
      s2 += st[d] * Wq2[d * 64 + cc];
    }
    tv[cc] = s1;
    qv[cc] = s2;
  }
  __syncthreads();

  const h16* K2 = K2h + (size_t)b * 128 * 64;
  const h16* V2 = V2h + (size_t)b * 128 * 64;
  for (int j = lane; j < 128; j += 32) {
    float s = 0.f;
    for (int d = 0; d < 64; ++d) s += qv[d] * (float)K2[(size_t)j * 64 + d];
    sc[j] = s * 0.125f;
  }
  __syncthreads();

  float mx = -1e30f;
  for (int j = 0; j < 128; ++j) mx = fmaxf(mx, sc[j]);
  float den = 0.f;
  for (int j = 0; j < 128; ++j) den += __expf(sc[j] - mx);
  float rden = 1.f / den;

  for (int cc = lane; cc < 64; cc += 32) {
    float s = 0.f;
    for (int j = 0; j < 128; ++j) s += __expf(sc[j] - mx) * (float)V2[(size_t)j * 64 + cc];
    ctx[cc] = s * rden;
  }
  __syncthreads();

  for (int cc = lane; cc < 64; cc += 32) {
    float s = bo1[cc] + bo2[cc];
    for (int d = 0; d < 64; ++d) s += tv[d] * Wo1[d * 64 + cc] + ctx[d] * Wo2[d * 64 + cc];
    outp[(size_t)b * 129 * 64 + cc] = s;  // row 0 = target
  }
}

// ---------------------------------------------------------------------------
extern "C" void kernel_launch(void* const* d_in, const int* in_sizes, int n_in,
                              void* d_out, int out_size, void* d_ws, size_t ws_size,
                              hipStream_t stream) {
  (void)in_sizes; (void)n_in; (void)out_size; (void)ws_size;
  const float* map_states   = (const float*)d_in[0];
  const float* agent_states = (const float*)d_in[1];
  const float* m_W1 = (const float*)d_in[2];
  const float* m_b1 = (const float*)d_in[3];
  const float* m_g1 = (const float*)d_in[4];
  const float* m_be1 = (const float*)d_in[5];
  const float* m_W2 = (const float*)d_in[6];
  const float* m_b2 = (const float*)d_in[7];
  const float* m_g2 = (const float*)d_in[8];
  const float* m_be2 = (const float*)d_in[9];
  const float* a_W1 = (const float*)d_in[10];
  const float* a_b1 = (const float*)d_in[11];
  const float* a_g1 = (const float*)d_in[12];
  const float* a_be1 = (const float*)d_in[13];
  const float* a_W2 = (const float*)d_in[14];
  const float* a_b2 = (const float*)d_in[15];
  const float* a_g2 = (const float*)d_in[16];
  const float* a_be2 = (const float*)d_in[17];
  const float* att_Wq = (const float*)d_in[18];  // [3][64][64]
  const float* att_bq = (const float*)d_in[19];  // [3][64]
  const float* att_Wk = (const float*)d_in[20];
  const float* att_bk = (const float*)d_in[21];
  const float* att_Wv = (const float*)d_in[22];
  const float* att_bv = (const float*)d_in[23];
  const float* att_Wo = (const float*)d_in[24];
  const float* att_bo = (const float*)d_in[25];
  float* outp = (float*)d_out;

  char* p = (char*)d_ws;
  float* mapEmb  = (float*)p; p += (size_t)32 * 128 * 64 * 4;
  float* mapEmb2 = (float*)p; p += (size_t)32 * 128 * 64 * 4;
  float* targetB = (float*)p; p += (size_t)32 * 64 * 4;
  h16* Qp  = (h16*)p; p += (size_t)32 * 16 * 512 * 2;  // 16 packed A tiles / b
  h16* Kp  = (h16*)p; p += (size_t)32 * 18 * 512 * 2;  // 18 packed B^T tiles / b
  h16* Vp  = (h16*)p; p += (size_t)32 * 20 * 512 * 2;  // 20 packed B tiles / b
  h16* m2h = (h16*)p; p += (size_t)32 * 128 * 64 * 2;
  h16* K2h = (h16*)p; p += (size_t)32 * 128 * 64 * 2;
  h16* V2h = (h16*)p; p += (size_t)32 * 128 * 64 * 2;

  // 1) map local graph: one wave per (b,m); input (20x16)
  k_local_graph<<<dim3(32 * 128), 32, 0, stream>>>(
      map_states, 20 * 16, 16, 20,
      m_W1, m_b1, m_g1, m_be1, m_W2, m_b2, m_g2, m_be2, mapEmb);

  // 2) agent local graph: only a=0 feeds the output; input (32x4)
  k_local_graph<<<dim3(32), 32, 0, stream>>>(
      agent_states, 32 * 32 * 4, 4, 32,
      a_W1, a_b1, a_g1, a_be1, a_W2, a_b2, a_g2, a_be2, targetB);

  // 3) attention-0 projections -> packed operand layouts
  k_proj0<<<dim3(32, 9), 32, 0, stream>>>(
      mapEmb, targetB,
      att_Wq, att_bq, att_Wk, att_bk, att_Wv, att_bv,
      Qp, Kp, Vp);

  // 4) attention-0 core + residual
  k_attn0<<<dim3(32, 8), 32, 0, stream>>>(
      Qp, Kp, Vp, att_Wo, att_bo, mapEmb, mapEmb2, m2h, outp);

  // 5) K2/V2 projections for t_m2a
  k_proj2<<<dim3(32, 8), 32, 0, stream>>>(
      m2h,
      att_Wk + 2 * 64 * 64, att_bk + 2 * 64,
      att_Wv + 2 * 64 * 64, att_bv + 2 * 64,
      K2h, V2h);

  // 6) target row (t_a + t_m2a)
  k_target<<<dim3(32), 32, 0, stream>>>(
      targetB, K2h, V2h,
      att_Wv + 1 * 64 * 64, att_bv + 1 * 64,
      att_Wo + 1 * 64 * 64, att_bo + 1 * 64,
      att_Wq + 2 * 64 * 64, att_bq + 2 * 64,
      att_Wo + 2 * 64 * 64, att_bo + 2 * 64,
      outp);
}
